// EdgeSAGE_17325898072720
// MI455X (gfx1250) — matrix operand order repaired and checked by reading, not compiled, not run on hardware
//
#include <hip/hip_runtime.h>

// ---------------------------------------------------------------------------
// EdgeSAGE on MI455X (gfx1250): scatter-mean SAGE x2 + edge MLP, all GEMMs via
// v_wmma_f32_16x16x32_bf16 (wave32 WMMA, f32 accumulate). Weights pre-shuffled
// into per-lane fragment order so B loads are 32B vector loads.
// ---------------------------------------------------------------------------

#define Nn 100000
#define Ee 1600000
#define EDGE_IN 16
#define Hd 128
#define KCAT 256          // [mean | x] concat K for SAGE layers (8 k-tiles)
#define KEDGE 288         // 272 padded up to 9 k-tiles of 32

typedef __attribute__((ext_vector_type(16))) __bf16 bf16x16;
typedef __attribute__((ext_vector_type(8)))  __bf16 bf16x8;
typedef __attribute__((ext_vector_type(8)))  float  floatx8;

// ---- WMMA fragment helpers (contiguous k order: lane half h holds k = 16h + c, same for A and B) -----------------

// A: 16x32 bf16 tile from LDS (row-major, row stride = `stride` elements).
// lane 0-15: rows 0-15, K base 0; lanes 16-31: same rows, K offset +16.
// Per-lane components are one contiguous 16-element run -> ds_load_b128 x2.
__device__ inline bf16x16 load_a_frag(const __bf16* lds, int stride, int k0, int lane) {
  int row = lane & 15;
  int kg  = (lane >> 4) * 16;
  bf16x16 a;
#pragma unroll
  for (int c = 0; c < 16; ++c) {
    int k = c + kg;
    a[c] = lds[row * stride + k0 + k];
  }
  return a;
}

__device__ inline floatx8 wmma_bf16(bf16x16 a, bf16x16 b, floatx8 c) {
  return __builtin_amdgcn_wmma_f32_16x16x32_bf16(false, a, false, b, (short)0, c,
                                                 false, false);
}

__device__ inline void store_bf8(__bf16* dst, float4 a, float4 b) {
  bf16x8 v;
  v[0] = (__bf16)a.x; v[1] = (__bf16)a.y; v[2] = (__bf16)a.z; v[3] = (__bf16)a.w;
  v[4] = (__bf16)b.x; v[5] = (__bf16)b.y; v[6] = (__bf16)b.z; v[7] = (__bf16)b.w;
  *(bf16x8*)dst = v;
}

// ---- weight packing: f32 row-major -> bf16 fragment order -----------------
// Fragment order: out[(((ntile*nkt)+kk)*32 + lane)*16 + c], where the lane's
// B-fragment component c holds W[kk*32 + (lane>>4)*16 + c][ntile*16 + (lane&15)].

__global__ void pack_cat_frag(const float* __restrict__ Wl,
                              const float* __restrict__ Wr,
                              __bf16* __restrict__ out) {
  int idx = blockIdx.x * blockDim.x + threadIdx.x;   // KCAT*Hd = 32768
  if (idx >= KCAT * Hd) return;
  int c    = idx & 15;
  int lane = (idx >> 4) & 31;
  int kk   = (idx >> 9) & 7;
  int wt   = idx >> 12;
  int col  = lane & 15;
  int kg   = (lane >> 4) * 16;
  int k = kk * 32 + kg + c;
  int n = wt * 16 + col;
  float v = (k < 128) ? Wl[k * Hd + n] : Wr[(k - 128) * Hd + n];
  out[idx] = (__bf16)v;
}

__global__ void pack_edge_frag(const float* __restrict__ Wm1,
                               __bf16* __restrict__ out) {
  int idx = blockIdx.x * blockDim.x + threadIdx.x;   // KEDGE*Hd = 36864
  if (idx >= KEDGE * Hd) return;
  int c    = idx & 15;
  int lane = (idx >> 4) & 31;
  int tmp  = idx >> 9;        // (wt*9 + kk)
  int kk   = tmp % 9;
  int wt   = tmp / 9;
  int col  = lane & 15;
  int kg   = (lane >> 4) * 16;
  int k = kk * 32 + kg + c;
  int n = wt * 16 + col;
  out[idx] = (k < 272) ? (__bf16)Wm1[k * Hd + n] : (__bf16)0.0f;
}

// ---- scatter-mean accumulation (f32 atomics) ------------------------------

__global__ void scatter_accum(const float* __restrict__ feat,
                              const int* __restrict__ src,
                              const int* __restrict__ dst,
                              float* __restrict__ msg,
                              float* __restrict__ cnt,
                              int addCnt) {
  long long idx = (long long)blockIdx.x * blockDim.x + threadIdx.x;
  if (idx >= (long long)Ee * 32) return;
  int e = (int)(idx >> 5);
  int c = (int)(idx & 31);
  int s = src[e], d = dst[e];
  float4 v = *(const float4*)(feat + (long long)s * Hd + c * 4);
  float* mp = msg + (long long)d * Hd + c * 4;
  atomicAdd(mp + 0, v.x);
  atomicAdd(mp + 1, v.y);
  atomicAdd(mp + 2, v.z);
  atomicAdd(mp + 3, v.w);
  if (addCnt && c == 0) atomicAdd(cnt + d, 1.0f);
}

// ---- fused SAGE layer: relu([mean|x] @ Wcat + b) --------------------------
// 256 threads = 8 waves; 16 node rows per block; wave w owns cols 16w..16w+15.

__global__ __launch_bounds__(256) void node_gemm_relu(
    const float* __restrict__ msg, const float* __restrict__ cnt,
    const float* __restrict__ xin, const __bf16* __restrict__ Wpack,
    const float* __restrict__ bias, float* __restrict__ out_f32,
    __bf16* __restrict__ out_bf) {
  __shared__ __bf16 As[16 * KCAT];
  int t = threadIdx.x;
  int row0 = blockIdx.x * 16;

  // stage A = [mean | x] as bf16, 8 elements (16B) per store.
  // 16 rows * 32 chunks = 512 chunks, 2 per thread.
#pragma unroll
  for (int j = 0; j < 2; ++j) {
    int ch = t + j * 256;
    int row = ch >> 5;
    int k8  = (ch & 31) * 8;
    int node = row0 + row;
    float4 a, b;
    if (k8 < 128) {
      float cdeg = cnt[node];
      cdeg = (cdeg > 1.0f) ? cdeg : 1.0f;
      float inv = 1.0f / cdeg;
      const float* sp = msg + (long long)node * Hd + k8;
      a = *(const float4*)(sp);
      b = *(const float4*)(sp + 4);
      a.x *= inv; a.y *= inv; a.z *= inv; a.w *= inv;
      b.x *= inv; b.y *= inv; b.z *= inv; b.w *= inv;
    } else {
      const float* sp = xin + (long long)node * Hd + (k8 - 128);
      a = *(const float4*)(sp);
      b = *(const float4*)(sp + 4);
    }
    store_bf8(&As[row * KCAT + k8], a, b);
  }
  __syncthreads();

  int wave = t >> 5, lane = t & 31;
  int n0 = wave * 16, col = lane & 15;

  // B fragments: contiguous 32B per lane (global_load_b128 x2 each)
  bf16x16 bfr[8];
#pragma unroll
  for (int kk = 0; kk < 8; ++kk) {
    const bf16x16* wp =
        (const bf16x16*)Wpack + ((size_t)(wave * 8 + kk) * 32 + lane);
    bfr[kk] = *wp;
  }

  floatx8 acc;
  float bv = bias[n0 + col];
#pragma unroll
  for (int r = 0; r < 8; ++r) acc[r] = bv;

#pragma unroll
  for (int kk = 0; kk < 8; ++kk) {
    bf16x16 a = load_a_frag(As, KCAT, kk * 32, lane);
    acc = wmma_bf16(a, bfr[kk], acc);
  }

  int rbase = (lane >> 4) * 8;
#pragma unroll
  for (int r = 0; r < 8; ++r) {
    int m = r + rbase;
    float v = acc[r];
    v = (v > 0.0f) ? v : 0.0f;
    long long o = (long long)(row0 + m) * Hd + n0 + col;
    if (out_f32) out_f32[o] = v;
    if (out_bf)  out_bf[o]  = (__bf16)v;
  }
}

// ---- edge MLP: relu([h2[src]|h2[dst]|ea] @ Wm1 + bm1) @ Wm2 + bm2 ---------
// 16 edges per 256-thread block; K = 288 (zero-padded).

__global__ __launch_bounds__(256) void edge_mlp(
    const __bf16* __restrict__ h2, const int* __restrict__ src,
    const int* __restrict__ dst, const float* __restrict__ eattr,
    const __bf16* __restrict__ Wm1p, const float* __restrict__ bm1,
    const float* __restrict__ Wm2, const float* __restrict__ bm2,
    float* __restrict__ out) {
  __shared__ __bf16 As[16 * KEDGE];
  __shared__ float  Ts[16 * Hd];
  __shared__ float  wm2s[Hd];
  __shared__ float  P[256];
  __shared__ int    sid[32];
  int t = threadIdx.x;
  int e0 = blockIdx.x * 16;

  if (t < 16)       sid[t] = src[e0 + t];
  else if (t < 32)  sid[t] = dst[e0 + (t - 16)];
  if (t >= 128 && t < 256) wm2s[t - 128] = Wm2[t - 128];
  __syncthreads();

  // stage 16x288 bf16 tile; 36 chunks of 8 per row, 576 chunks total.
#pragma unroll
  for (int j = 0; j < 3; ++j) {
    int ch = t + j * 256;
    if (ch < 576) {
      int row = ch / 36;
      int q   = ch % 36;
      int k8  = q * 8;
      __bf16* dp = &As[row * KEDGE + k8];
      if (q < 16) {                 // h2[src], bf16 -> direct 16B copy
        *(uint4*)dp = *(const uint4*)(h2 + (long long)sid[row] * Hd + k8);
      } else if (q < 32) {          // h2[dst]
        *(uint4*)dp =
            *(const uint4*)(h2 + (long long)sid[16 + row] * Hd + (k8 - 128));
      } else if (q < 34) {          // edge_attr (16 cols -> chunks 32,33)
        const float* sp = eattr + (long long)(e0 + row) * EDGE_IN + (k8 - 256);
        store_bf8(dp, *(const float4*)sp, *(const float4*)(sp + 4));
      } else {                      // zero padding (K 272..287)
        uint4 z = {0, 0, 0, 0};
        *(uint4*)dp = z;
      }
    }
  }
  __syncthreads();

  int wave = t >> 5, lane = t & 31;
  int n0 = wave * 16, col = lane & 15;

  bf16x16 bfr[9];
#pragma unroll
  for (int kk = 0; kk < 9; ++kk) {
    const bf16x16* wp =
        (const bf16x16*)Wm1p + ((size_t)(wave * 9 + kk) * 32 + lane);
    bfr[kk] = *wp;
  }

  floatx8 acc;
  float bv = bm1[n0 + col];
#pragma unroll
  for (int r = 0; r < 8; ++r) acc[r] = bv;

#pragma unroll
  for (int kk = 0; kk < 9; ++kk) {
    bf16x16 a = load_a_frag(As, KEDGE, kk * 32, lane);
    acc = wmma_bf16(a, bfr[kk], acc);
  }

  int rbase = (lane >> 4) * 8;
#pragma unroll
  for (int r = 0; r < 8; ++r) {
    float v = acc[r];
    Ts[(r + rbase) * Hd + n0 + col] = (v > 0.0f) ? v : 0.0f;
  }
  __syncthreads();

  // head: out[e] = Ts[e,:] . Wm2 + bm2; 16 threads/edge, 8 elems each.
  {
    int e   = t >> 4;
    int seg = (t & 15) * 8;
    float s = 0.0f;
#pragma unroll
    for (int k = 0; k < 8; ++k) s += Ts[e * Hd + seg + k] * wm2s[seg + k];
    P[t] = s;
  }
  __syncthreads();
  if (t < 16) {
    float s = bm2[0];
#pragma unroll
    for (int k = 0; k < 16; ++k) s += P[t * 16 + k];
    out[e0 + t] = s;
  }
}

// ---------------------------------------------------------------------------

extern "C" void kernel_launch(void* const* d_in, const int* in_sizes, int n_in,
                              void* d_out, int out_size, void* d_ws, size_t ws_size,
                              hipStream_t stream) {
  const float* x     = (const float*)d_in[0];
  const int*   ei    = (const int*)d_in[1];
  const float* eattr = (const float*)d_in[2];
  const float* Wl1   = (const float*)d_in[3];
  const float* Wr1   = (const float*)d_in[4];
  const float* b1    = (const float*)d_in[5];
  const float* Wl2   = (const float*)d_in[6];
  const float* Wr2   = (const float*)d_in[7];
  const float* b2    = (const float*)d_in[8];
  const float* Wm1   = (const float*)d_in[9];
  const float* bm1   = (const float*)d_in[10];
  const float* Wm2   = (const float*)d_in[11];
  const float* bm2   = (const float*)d_in[12];
  const int* srcIdx = ei;
  const int* dstIdx = ei + Ee;
  float* out = (float*)d_out;

  // workspace carve-up
  char* p = (char*)d_ws;
  auto take = [&](size_t bytes) {
    char* r = p;
    p += (bytes + 255) & ~(size_t)255;
    return r;
  };
  __bf16* wcat1 = (__bf16*)take((size_t)KCAT * Hd * 2);
  __bf16* wcat2 = (__bf16*)take((size_t)KCAT * Hd * 2);
  __bf16* wm1p  = (__bf16*)take((size_t)KEDGE * Hd * 2);
  float*  msg   = (float*)take((size_t)Nn * Hd * 4);
  float*  cnt   = (float*)take((size_t)Nn * 4);
  float*  h1    = (float*)take((size_t)Nn * Hd * 4);
  __bf16* h2bf  = (__bf16*)take((size_t)Nn * Hd * 2);

  // 1) pack weights to bf16 fragment order
  pack_cat_frag<<<(KCAT * Hd + 255) / 256, 256, 0, stream>>>(Wl1, Wr1, wcat1);
  pack_cat_frag<<<(KCAT * Hd + 255) / 256, 256, 0, stream>>>(Wl2, Wr2, wcat2);
  pack_edge_frag<<<(KEDGE * Hd + 255) / 256, 256, 0, stream>>>(Wm1, wm1p);

  // 2) layer 1: scatter-mean + fused GEMM (relu)
  hipMemsetAsync(msg, 0, (size_t)Nn * Hd * 4, stream);
  hipMemsetAsync(cnt, 0, (size_t)Nn * 4, stream);
  {
    long long tot = (long long)Ee * 32;
    scatter_accum<<<(unsigned)((tot + 255) / 256), 256, 0, stream>>>(
        x, srcIdx, dstIdx, msg, cnt, 1);
  }
  node_gemm_relu<<<Nn / 16, 256, 0, stream>>>(msg, cnt, x, wcat1, b1, h1,
                                              (__bf16*)nullptr);

  // 3) layer 2 (degree counts unchanged)
  hipMemsetAsync(msg, 0, (size_t)Nn * Hd * 4, stream);
  {
    long long tot = (long long)Ee * 32;
    scatter_accum<<<(unsigned)((tot + 255) / 256), 256, 0, stream>>>(
        h1, srcIdx, dstIdx, msg, cnt, 0);
  }
  node_gemm_relu<<<Nn / 16, 256, 0, stream>>>(msg, cnt, h1, wcat2, b2,
                                              (float*)nullptr, h2bf);

  // 4) edge MLP head
  edge_mlp<<<Ee / 16, 256, 0, stream>>>(h2bf, srcIdx, dstIdx, eattr, wm1p, bm1,
                                        Wm2, bm2, out);
}
